// Arch7V4GraphEncoder_33243046871801
// MI455X (gfx1250) — matrix operand order repaired and checked by reading, not compiled, not run on hardware
//
#include <hip/hip_runtime.h>

typedef __attribute__((ext_vector_type(16))) _Float16 v16h;
typedef __attribute__((ext_vector_type(4)))  _Float16 v4h;
typedef __attribute__((ext_vector_type(8)))  float    v8f;

#define H 128

// ---------------------------------------------------------------------------
// Fused GEMM: Y[M,128] = post( (X[+Xadd])[gather] @ W^T + b )
// W is [128,128] row-major with Y[m][n] = sum_k X[m][k]*W[n][k].
// Compile-time variants:
//   ADD    : X := X + Xadd (elementwise, same rows)   [fuses z = h + agg]
//   GATHER : input rows taken from gatherIdx[row]     [fuses vv/kk gathers]
//   BN     : v = v*scale[n] + shift[n] after bias
//   RELU   : v = max(v, 0)
//   VALID  : zero rows whose validIds[row] < 0        [vf mask]
//   ACC    : Y += v instead of Y = v                  [partial-sum fusion]
// One wave computes one 16-row tile across all 8 column tiles using
// v_wmma_f32_16x16x32_f16; weights staged in LDS as f16 (32 KB).
// ---------------------------------------------------------------------------
template <bool ADD, bool GATHER, bool BN, bool RELU, bool VALID, bool ACC>
__global__ __launch_bounds__(256)
void gemm128(const float* __restrict__ X, const float* __restrict__ Xadd,
             const int* __restrict__ gatherIdx,
             const float* __restrict__ W, const float* __restrict__ bias,
             const float* __restrict__ scale, const float* __restrict__ shift,
             const int* __restrict__ validIds,
             float* __restrict__ Y, int Mtiles)
{
    __shared__ __align__(32) _Float16 Wl[H * H];
    // vectorized f32 -> f16 staging: 4096 float4 across 256 threads
    for (int i = threadIdx.x; i < (H * H) / 4; i += 256) {
        float4 f = ((const float4*)W)[i];
        v4h hv;
        hv[0] = (_Float16)f.x; hv[1] = (_Float16)f.y;
        hv[2] = (_Float16)f.z; hv[3] = (_Float16)f.w;
        *(v4h*)(Wl + i * 4) = hv;
    }
    __syncthreads();

    const int wave = threadIdx.x >> 5;
    const int lane = threadIdx.x & 31;
    const int tile = blockIdx.x * 8 + wave;
    if (tile >= Mtiles) return;              // wave-uniform: EXEC stays full

    const int half = lane >> 4;              // 0: lanes 0-15, 1: lanes 16-31
    const int mloc = lane & 15;
    const int rowBase = tile * 16;

    const int srcRow = rowBase + mloc;
    int row = srcRow;
    if (GATHER) row = gatherIdx[srcRow];
    const float* xr  = X + row * H;
    const float* xr2 = ADD ? (Xadd + srcRow * H) : nullptr;

    // Preload per-ct epilogue constants (lane-private, ct-indexed).
    float bias8[8];
#pragma unroll
    for (int ct = 0; ct < 8; ++ct) bias8[ct] = bias[ct * 16 + mloc];
    float sc8[8], sh8[8];
    if (BN) {
#pragma unroll
        for (int ct = 0; ct < 8; ++ct) {
            sc8[ct] = scale[ct * 16 + mloc];
            sh8[ct] = shift[ct * 16 + mloc];
        }
    }
    // validity mask is ct-invariant: load once per wave (8 rows per lane half)
    float vf8[8];
    if (VALID) {
#pragma unroll
        for (int r = 0; r < 8; ++r)
            vf8[r] = (validIds[rowBase + half * 8 + r] >= 0) ? 1.0f : 0.0f;
    }

    // A strip: 16 rows x 128 k, contiguous k order (same order as the B loads).
    // lane l, elem e -> X[m = l&15][k = ks*32 + 16*(l>=16) + e]
    v16h a[4];
#pragma unroll
    for (int ks = 0; ks < 4; ++ks) {
#pragma unroll
        for (int run = 0; run < 2; ++run) {
            const int kb = ks * 32 + half * 16 + run * 8;
            float4 f0 = *(const float4*)(xr + kb);
            float4 f1 = *(const float4*)(xr + kb + 4);
            if (ADD) {
                float4 g0 = *(const float4*)(xr2 + kb);
                float4 g1 = *(const float4*)(xr2 + kb + 4);
                f0.x += g0.x; f0.y += g0.y; f0.z += g0.z; f0.w += g0.w;
                f1.x += g1.x; f1.y += g1.y; f1.z += g1.z; f1.w += g1.w;
            }
            const int e = run * 8;
            a[ks][e + 0] = (_Float16)f0.x; a[ks][e + 1] = (_Float16)f0.y;
            a[ks][e + 2] = (_Float16)f0.z; a[ks][e + 3] = (_Float16)f0.w;
            a[ks][e + 4] = (_Float16)f1.x; a[ks][e + 5] = (_Float16)f1.y;
            a[ks][e + 6] = (_Float16)f1.z; a[ks][e + 7] = (_Float16)f1.w;
        }
    }

#pragma unroll 1
    for (int ct = 0; ct < 8; ++ct) {
        const int n = ct * 16 + mloc;
        // Load all 4 B tiles first (8x ds_load_b128, single wait), then run
        // the WMMA chain without intervening LDS waits.
        v16h bt[4];
#pragma unroll
        for (int ks = 0; ks < 4; ++ks)
            bt[ks] = *(const v16h*)(Wl + n * H + ks * 32 + half * 16);
        v8f acc = {};
#pragma unroll
        for (int ks = 0; ks < 4; ++ks)
            acc = __builtin_amdgcn_wmma_f32_16x16x32_f16(
                false, a[ks], false, bt[ks], (short)0, acc, false, false);

        const float bn = bias8[ct];
#pragma unroll
        for (int r = 0; r < 8; ++r) {
            const int mrow = rowBase + half * 8 + r;
            float v = acc[r] + bn;
            if (BN)    v = v * sc8[ct] + sh8[ct];
            if (RELU)  v = fmaxf(v, 0.0f);
            if (VALID) v *= vf8[r];
            float* op = Y + mrow * H + ct * 16 + mloc;
            if (ACC)   v += *op;
            *op = v;
        }
    }
}

// ---------------------------------------------------------------------------
// Elementwise / scatter kernels (one row per 32 threads, float4 per thread)
// ---------------------------------------------------------------------------
__global__ __launch_bounds__(256)
void k_setup_idx(const int* __restrict__ node_ids, const int* __restrict__ ksub,
                 int SK, int* __restrict__ clamped, int* __restrict__ rof)
{
    int i = blockIdx.x * 256 + threadIdx.x;
    if (i >= SK) return;
    int nid = node_ids[i];
    clamped[i] = nid > 0 ? nid : 0;
    int k = ksub[0];
    rof[i] = (i / k) * k;
}

__global__ __launch_bounds__(256)
void k_embed(const float* __restrict__ atom_table, const float* __restrict__ role_table,
             const int* __restrict__ atom_ids, const int* __restrict__ node_ids,
             const int* __restrict__ ksub, int SK, float* __restrict__ h)
{
    int t = blockIdx.x * 256 + threadIdx.x;
    int i = t >> 5; if (i >= SK) return;
    int f = (t & 31) * 4;
    int k = ksub[0];
    int role = ((i % k) == 0) ? 1 : 0;
    float vf = (node_ids[i] >= 0) ? 1.0f : 0.0f;
    const float* ar = atom_table + atom_ids[i] * H + f;
    const float* rr = role_table + role * H + f;
    float* hp = h + i * H + f;
#pragma unroll
    for (int q = 0; q < 4; ++q) hp[q] = (ar[q] + rr[q]) * vf;
}

// msg = relu(x[src] + bond_table[bond_id]); agg[tgt] += msg
__global__ __launch_bounds__(256)
void k_scatter_edges(const float* __restrict__ x, const float* __restrict__ bond_table,
                     const int* __restrict__ bond_ids, const int* __restrict__ ei,
                     int E, float* __restrict__ agg)
{
    int t = blockIdx.x * 256 + threadIdx.x;
    int e = t >> 5; if (e >= E) return;
    int f = (t & 31) * 4;
    int src = ei[e], tgt = ei[E + e];
    const float* xp = x + src * H + f;
    const float* bp = bond_table + bond_ids[e] * H + f;
    float* ap = agg + tgt * H + f;
#pragma unroll
    for (int q = 0; q < 4; ++q) {
        float m = xp[q] + bp[q];
        if (m > 0.0f) atomicAdd(ap + q, m);   // relu: skip non-positive
    }
}

__global__ __launch_bounds__(256)
void k_scatter_rows(const float* __restrict__ x, const int* __restrict__ node_ids,
                    int SK, float* __restrict__ sums, float* __restrict__ cnt)
{
    int t = blockIdx.x * 256 + threadIdx.x;
    int i = t >> 5; if (i >= SK) return;
    int nid = node_ids[i]; if (nid < 0) return;
    int f = (t & 31) * 4;
    const float* xp = x + i * H + f;
    float* sp = sums + nid * H + f;
#pragma unroll
    for (int q = 0; q < 4; ++q) atomicAdd(sp + q, xp[q]);
    if (f == 0) atomicAdd(cnt + nid, 1.0f);
}

__global__ __launch_bounds__(256)
void k_scatter_roots(const float* __restrict__ x, const int* __restrict__ node_ids,
                     const int* __restrict__ ksub, int SK,
                     float* __restrict__ sums, float* __restrict__ cnt)
{
    int t = blockIdx.x * 256 + threadIdx.x;
    int i = t >> 5; if (i >= SK) return;
    int k = ksub[0]; if (i % k) return;       // roots only
    int nid = node_ids[i]; if (nid < 0) return;
    int f = (t & 31) * 4;
    const float* xp = x + i * H + f;
    float* sp = sums + nid * H + f;
#pragma unroll
    for (int q = 0; q < 4; ++q) atomicAdd(sp + q, xp[q]);
    if (f == 0) atomicAdd(cnt + nid, 1.0f);
}

__global__ __launch_bounds__(256)
void k_div(float* __restrict__ sums, const float* __restrict__ cnt, int N)
{
    int t = blockIdx.x * 256 + threadIdx.x;
    int n = t >> 5; if (n >= N) return;
    int f = (t & 31) * 4;
    float c = cnt[n]; c = c > 1.0f ? c : 1.0f;
    float inv = 1.0f / c;
    float* sp = sums + n * H + f;
#pragma unroll
    for (int q = 0; q < 4; ++q) sp[q] *= inv;
}

// h = relu(A + h2n[clamped]) * vf
__global__ __launch_bounds__(256)
void k_combine(const float* __restrict__ A, const float* __restrict__ h2n,
               const int* __restrict__ clamped, const int* __restrict__ node_ids,
               int SK, float* __restrict__ h)
{
    int t = blockIdx.x * 256 + threadIdx.x;
    int i = t >> 5; if (i >= SK) return;
    int f = (t & 31) * 4;
    float* hp = h + i * H + f;
    if (node_ids[i] < 0) {
#pragma unroll
        for (int q = 0; q < 4; ++q) hp[q] = 0.0f;
        return;
    }
    const float* ap = A + i * H + f;
    const float* gp = h2n + clamped[i] * H + f;
#pragma unroll
    for (int q = 0; q < 4; ++q) {
        float v = ap[q] + gp[q];
        hp[q] = v > 0.0f ? v : 0.0f;
    }
}

__global__ __launch_bounds__(256)
void k_pool(const float* __restrict__ ne, const int* __restrict__ batch,
            int N, float* __restrict__ out)
{
    int t = blockIdx.x * 256 + threadIdx.x;
    int n = t >> 5; if (n >= N) return;
    int f = (t & 31) * 4;
    int g = batch[n];
    const float* np = ne + n * H + f;
    float* op = out + g * H + f;
#pragma unroll
    for (int q = 0; q < 4; ++q) atomicAdd(op + q, np[q]);
}

// ---------------------------------------------------------------------------
extern "C" void kernel_launch(void* const* d_in, const int* in_sizes, int n_in,
                              void* d_out, int out_size, void* d_ws, size_t ws_size,
                              hipStream_t stream)
{
    const float* atom_table = (const float*)d_in[0];
    const float* bond_table = (const float*)d_in[1];
    const float* role_table = (const float*)d_in[2];
    const float* lW1  = (const float*)d_in[3];
    const float* lb1  = (const float*)d_in[4];
    const float* lW2  = (const float*)d_in[5];
    const float* lb2  = (const float*)d_in[6];
    const float* gW1  = (const float*)d_in[7];
    const float* gb1  = (const float*)d_in[8];
    const float* gW2  = (const float*)d_in[9];
    const float* gb2  = (const float*)d_in[10];
    const float* lbn_g = (const float*)d_in[11];
    const float* lbn_b = (const float*)d_in[12];
    const float* gbn_g = (const float*)d_in[13];
    const float* gbn_b = (const float*)d_in[14];
    const float* skipW = (const float*)d_in[15];
    const float* skipb = (const float*)d_in[16];
    const float* vvW   = (const float*)d_in[17];
    const float* vvb   = (const float*)d_in[18];
    const float* kkW   = (const float*)d_in[19];
    const float* kkb   = (const float*)d_in[20];
    const int* atom_ids        = (const int*)d_in[21];
    const int* bond_ids_intra  = (const int*)d_in[22];
    const int* bond_ids_global = (const int*)d_in[23];
    const int* intra_ei        = (const int*)d_in[24];
    const int* node_ids        = (const int*)d_in[25];
    const int* edge_index      = (const int*)d_in[26];
    const int* batch           = (const int*)d_in[28];
    const int* ksub            = (const int*)d_in[29];

    const int SK = in_sizes[21];
    const int EI = in_sizes[22];
    const int EG = in_sizes[23];
    const int N  = in_sizes[28];
    const int L  = in_sizes[4] / H;

    // workspace carve-up
    char* ws = (char*)d_ws; size_t o = 0;
    auto carve = [&](size_t bytes) -> void* {
        void* p = ws + o; o = (o + bytes + 255) & ~(size_t)255; return p;
    };
    const size_t skB = (size_t)SK * H * sizeof(float);
    const size_t nB  = (size_t)N  * H * sizeof(float);
    float* h   = (float*)carve(skB);   // node features
    float* A   = (float*)carve(skB);   // accumulator: skip + h1 + vv + kk
    float* B   = (float*)carve(skB);   // intra agg / scratch
    float* C   = (float*)carve(skB);   // hmid / hmidN / scratch
    float* D   = (float*)carve(nB);    // x_sum / x_vv_c / node_embs
    float* E   = (float*)carve(nB);    // aggN / h2n
    float* cnt = (float*)carve((size_t)N * sizeof(float));
    int* clamped = (int*)carve((size_t)SK * sizeof(int));
    int* rof     = (int*)carve((size_t)SK * sizeof(int));
    (void)ws_size; (void)n_in;

    const int gSK = (SK * 32 + 255) / 256;
    const int gN  = (N  * 32 + 255) / 256;
    const int gEI = (EI * 32 + 255) / 256;
    const int gEG = (EG * 32 + 255) / 256;
    const int tSK = SK / 16, tN = N / 16;
    const int bSK = (tSK + 7) / 8, bN = (tN + 7) / 8;

    k_setup_idx<<<(SK + 255) / 256, 256, 0, stream>>>(node_ids, ksub, SK, clamped, rof);
    k_embed<<<gSK, 256, 0, stream>>>(atom_table, role_table, atom_ids, node_ids, ksub, SK, h);

    for (int l = 0; l < L; ++l) {
        const size_t w = (size_t)l * H * H, v = (size_t)l * H;
        // 1) A = (h @ skipW^T + b) * vf                    <VALID>
        gemm128<false,false,false,false,true,false><<<bSK, 256, 0, stream>>>(
            h, nullptr, nullptr, skipW + w, skipb + v,
            nullptr, nullptr, node_ids, A, tSK);
        // 2) intra aggregation
        hipMemsetAsync(B, 0, skB, stream);
        k_scatter_edges<<<gEI, 256, 0, stream>>>(h, bond_table, bond_ids_intra,
                                                 intra_ei, EI, B);
        // 3) hmid = relu((h+agg) @ lW1^T + b1)             <ADD, RELU>
        gemm128<true,false,false,true,false,false><<<bSK, 256, 0, stream>>>(
            h, B, nullptr, lW1 + w, lb1 + v,
            nullptr, nullptr, nullptr, C, tSK);
        // 4) A += ((hmid @ lW2^T + b2)*bn_g + bn_b) * vf   <BN, VALID, ACC>
        gemm128<false,false,true,false,true,true><<<bSK, 256, 0, stream>>>(
            C, nullptr, nullptr, lW2 + w, lb2 + v,
            lbn_g + v, lbn_b + v, node_ids, A, tSK);
        // 5) x_sum = seg_mean(h, safe, N)
        hipMemsetAsync(D, 0, nB, stream);
        hipMemsetAsync(cnt, 0, (size_t)N * sizeof(float), stream);
        k_scatter_rows<<<gSK, 256, 0, stream>>>(h, node_ids, SK, D, cnt);
        k_div<<<gN, 256, 0, stream>>>(D, cnt, N);
        // 6) global aggregation
        hipMemsetAsync(E, 0, nB, stream);
        k_scatter_edges<<<gEG, 256, 0, stream>>>(D, bond_table, bond_ids_global,
                                                 edge_index, EG, E);
        // 7) hmidN = relu((x_sum+aggN) @ gW1^T + b1)       <ADD, RELU>
        gemm128<true,false,false,true,false,false><<<bN, 256, 0, stream>>>(
            D, E, nullptr, gW1 + w, gb1 + v,
            nullptr, nullptr, nullptr, C, tN);
        // 8) h2n = (hmidN @ gW2^T + b2)*gbn_g + gbn_b      <BN>
        gemm128<false,false,true,false,false,false><<<bN, 256, 0, stream>>>(
            C, nullptr, nullptr, gW2 + w, gb2 + v,
            gbn_g + v, gbn_b + v, nullptr, E, tN);
        // 9) x_vv_c = seg_mean(h[roots], root_safe, N)
        hipMemsetAsync(D, 0, nB, stream);
        hipMemsetAsync(cnt, 0, (size_t)N * sizeof(float), stream);
        k_scatter_roots<<<gSK, 256, 0, stream>>>(h, node_ids, ksub, SK, D, cnt);
        k_div<<<gN, 256, 0, stream>>>(D, cnt, N);
        // 10) A += (x_vv_c[clamped] @ vvW^T + b) * vf      <GATHER, VALID, ACC>
        gemm128<false,true,false,false,true,true><<<bSK, 256, 0, stream>>>(
            D, nullptr, clamped, vvW + w, vvb + v,
            nullptr, nullptr, node_ids, A, tSK);
        // 11) A += (h[root_of_flat] @ kkW^T + b) * vf      <GATHER, VALID, ACC>
        gemm128<false,true,false,false,true,true><<<bSK, 256, 0, stream>>>(
            h, nullptr, rof, kkW + w, kkb + v,
            nullptr, nullptr, node_ids, A, tSK);
        // 12) h = relu(A + h2n[clamped]) * vf
        k_combine<<<gSK, 256, 0, stream>>>(A, E, clamped, node_ids, SK, h);
    }

    // node_embs = seg_mean(h); out = segment_sum(node_embs, batch)
    hipMemsetAsync(D, 0, nB, stream);
    hipMemsetAsync(cnt, 0, (size_t)N * sizeof(float), stream);
    k_scatter_rows<<<gSK, 256, 0, stream>>>(h, node_ids, SK, D, cnt);
    k_div<<<gN, 256, 0, stream>>>(D, cnt, N);
    hipMemsetAsync(d_out, 0, (size_t)out_size * sizeof(float), stream);
    k_pool<<<gN, 256, 0, stream>>>(D, batch, N, (float*)d_out);
}
